// BigramLanguageModel_17325898072452
// MI455X (gfx1250) — compile-verified, matrix operand order checked by reading, not run on hardware
//
#include <hip/hip_runtime.h>
#include <math.h>

// ---- model dims (match reference) ----
#define V_    50257
#define E_    384
#define L_    8
#define H_    8
#define HS_   48
#define RD_   32
#define KVR_  64
#define QR_   64
#define HID_  1024
#define B_    8
#define T_    512
#define BT_   (B_*T_)

typedef __bf16 bf16;
typedef __attribute__((ext_vector_type(16))) __bf16 v16bf;
typedef __attribute__((ext_vector_type(8)))  float  v8f;

struct __align__(16) frag2 { uint4 lo; uint4 hi; };   // 32B = one v16bf fragment

// ---------------------------------------------------------------------------
// f32 -> bf16 conversion (no transpose; used for tok_emb which is already N x K)
// ---------------------------------------------------------------------------
__global__ void f32_to_bf16_kernel(const float* __restrict__ in, bf16* __restrict__ out, long n) {
  long i = (long)blockIdx.x * blockDim.x + threadIdx.x;
  if (i < n) out[i] = (bf16)in[i];
}

// ---------------------------------------------------------------------------
// Weight convert + transpose: in [z][K][N] f32 -> out [z][N][K] bf16.
// 32x32 LDS tile, coalesced reads and writes. K, N multiples of 32.
// ---------------------------------------------------------------------------
__global__ void transpose_f32_to_bf16_kernel(const float* __restrict__ in, bf16* __restrict__ out,
                                             int K, int N) {
  __shared__ float tile[32][33];
  const float* inz = in + (size_t)blockIdx.z * K * N;
  bf16* outz = out + (size_t)blockIdx.z * K * N;
  int k0 = blockIdx.y * 32, n0 = blockIdx.x * 32;
  int lane = threadIdx.x & 31, wv = threadIdx.x >> 5;
  #pragma unroll
  for (int p = 0; p < 4; ++p) {
    int r = wv + p * 8;
    tile[r][lane] = inz[(size_t)(k0 + r) * N + n0 + lane];
  }
  __syncthreads();
  #pragma unroll
  for (int p = 0; p < 4; ++p) {
    int r = wv + p * 8;
    outz[(size_t)(n0 + r) * K + k0 + lane] = (bf16)tile[lane][r];
  }
}

// ---------------------------------------------------------------------------
// token + position embedding: x[bt, e] = tok[idx[bt], e] + pos[t, e]
// ---------------------------------------------------------------------------
__global__ void embed_kernel(const int* __restrict__ idx, const float* __restrict__ tok,
                             const float* __restrict__ pos, float* __restrict__ x) {
  int i = blockIdx.x * blockDim.x + threadIdx.x;
  if (i >= BT_ * E_) return;
  int e = i % E_;
  int bt = i / E_;
  int t = bt % T_;
  x[i] = tok[(size_t)idx[bt] * E_ + e] + pos[(size_t)t * E_ + e];
}

// ---------------------------------------------------------------------------
// LayerNorm over E=384, one row per block, bf16 output for WMMA consumers
// ---------------------------------------------------------------------------
__global__ void ln_kernel(const float* __restrict__ x, const float* __restrict__ g,
                          const float* __restrict__ b, bf16* __restrict__ out) {
  int row = blockIdx.x, tid = threadIdx.x;
  const float* xr = x + (size_t)row * E_;
  __shared__ float red[256];
  float s = 0.f;
  for (int j = tid; j < E_; j += 256) s += xr[j];
  red[tid] = s; __syncthreads();
  for (int o = 128; o > 0; o >>= 1) { if (tid < o) red[tid] += red[tid + o]; __syncthreads(); }
  float mean = red[0] * (1.f / E_);
  __syncthreads();
  float s2 = 0.f;
  for (int j = tid; j < E_; j += 256) { float d = xr[j] - mean; s2 += d * d; }
  red[tid] = s2; __syncthreads();
  for (int o = 128; o > 0; o >>= 1) { if (tid < o) red[tid] += red[tid + o]; __syncthreads(); }
  float inv = rsqrtf(red[0] * (1.f / E_) + 1e-5f);
  for (int j = tid; j < E_; j += 256)
    out[(size_t)row * E_ + j] = (bf16)((xr[j] - mean) * inv * g[j] + b[j]);
}

// ---------------------------------------------------------------------------
// WMMA GEMM: C[M,N](f32) = A[M,K](bf16,row-major) @ B^T, with B given as
// Bt[N,K](bf16,row-major) -- all weights are stored pre-transposed so BOTH
// operands stage with contiguous 16B loads along K.
//   ACC=true: C += A@B via non-returning global_atomic_add_f32.
//
// Tiles staged in LDS in *fragment order* so each lane reads its whole 32B
// WMMA operand with two ds_load_b128s (no scalar LDS gathers).
// Mapping (ISA 7.12.2, 16-bit A 16x32): lane-half kb = 8*(lane>=16),
// element e -> k = kb + e + 8*(e>=8).  A run of 8 K at fixed M/N therefore
// lands in ONE lane at contiguous elements -> staged with one ds_store_b128.
// Fragment stride 24 elems (48B): 16B-aligned stores, bank-friendly reads.
//
// Out-of-range N columns are staged from a CLAMPED row (branch-free): a B
// column only affects its own output column, and the epilogue guard drops
// columns >= N, so duplicate data there is harmless.
//
// Block tile 64x64, BK=64 (2 WMMA K-steps per barrier), 8 waves, each wave
// owns a 16x32 output strip -> 4 v_wmma_f32_16x16x32_bf16 per stage.
// ---------------------------------------------------------------------------
template<bool ACC>
__global__ void gemm_wmma(const bf16* __restrict__ A, const bf16* __restrict__ Bt,
                          float* __restrict__ C, int M, int N, int K) {
  __shared__ __align__(16) bf16 AsF[2][4][32][24];   // [ks][m-subtile][lane][frag]
  __shared__ __align__(16) bf16 BsF[2][4][32][24];   // [ks][n-subtile][lane][frag]
  int m0 = blockIdx.y * 64, n0 = blockIdx.x * 64;
  int tid = threadIdx.x, w = tid >> 5, lane = tid & 31;
  int wms = w & 3;          // A (m) subtile for this wave
  int wns = (w >> 2) * 2;   // B (n) subtile base for this wave
  v8f acc0 = {}; v8f acc1 = {};

  // loop-invariant staging coordinates
  int ar = tid >> 2;                      // A tile row 0..63
  int nl = tid & 63;                      // B tile col 0..63
  int nc = n0 + nl; if (nc > N - 1) nc = N - 1;   // clamped (branch-free OOB)
  const bf16* Arow = A  + (size_t)(m0 + ar) * K;
  const bf16* Brow = Bt + (size_t)nc * K;

  for (int k0 = 0; k0 < K; k0 += 64) {
    // ---- stage A tile 64x64: 16B vector loads -> fragment-order stores ----
    {
      int mf = ar & 15, s = ar >> 4;
      #pragma unroll
      for (int p = 0; p < 2; ++p) {
        int kq = (tid & 3) * 8 + p * 32;          // 0,8,16,24 / 32,...,56
        int ks = kq >> 5, kk = kq & 31;
        uint4 v = *(const uint4*)(Arow + k0 + kq);
        *(uint4*)&AsF[ks][s][mf + 16 * ((kk >> 3) & 1)][8 * (kk >> 4)] = v;
      }
    }
    // ---- stage B tile 64x64: Bt rows are K-contiguous -> same trick ----
    {
      int s = nl >> 4, nf = nl & 15;
      #pragma unroll
      for (int p = 0; p < 2; ++p) {
        int kc = (tid >> 6) + p * 4;              // 0..7 (8 K-values each)
        int ks = kc >> 2, kcc = kc & 3;
        uint4 v = *(const uint4*)(Brow + k0 + kc * 8);
        *(uint4*)&BsF[ks][s][nf + 16 * (kcc & 1)][(kcc >> 1) * 8] = v;
      }
    }
    if (k0 + 64 < K) {  // gfx1250 global_prefetch_b8 of next tiles (near-temporal)
      __builtin_prefetch(Arow + (k0 + 64) + (tid & 3) * 8, 0, 3);
      __builtin_prefetch(Brow + (k0 + 64) + (tid >> 6) * 8, 0, 3);
    }
    __syncthreads();

    #pragma unroll
    for (int ks = 0; ks < 2; ++ks) {
      frag2 fa  = *(const frag2*)&AsF[ks][wms][lane][0];
      frag2 fb0 = *(const frag2*)&BsF[ks][wns][lane][0];
      frag2 fb1 = *(const frag2*)&BsF[ks][wns + 1][lane][0];
      v16bf a  = __builtin_bit_cast(v16bf, fa);
      v16bf b0 = __builtin_bit_cast(v16bf, fb0);
      v16bf b1 = __builtin_bit_cast(v16bf, fb1);
      acc0 = __builtin_amdgcn_wmma_f32_16x16x32_bf16(false, a, false, b0, (short)0, acc0, false, false);
      acc1 = __builtin_amdgcn_wmma_f32_16x16x32_bf16(false, a, false, b1, (short)0, acc1, false, false);
    }
    __syncthreads();
  }

  // ---- epilogue: C layout (7.12.2): n = lane&15, vgpr v -> m = v + 8*(lane>=16)
  int rbase = m0 + wms * 16 + 8 * (lane >> 4);
  int c0 = n0 + wns * 16 + (lane & 15);
  int c1 = c0 + 16;
  #pragma unroll
  for (int v = 0; v < 8; ++v) {
    size_t r = (size_t)(rbase + v);
    if (c0 < N) { if (ACC) atomicAdd(&C[r * N + c0], acc0[v]); else C[r * N + c0] = acc0[v]; }
    if (c1 < N) { if (ACC) atomicAdd(&C[r * N + c1], acc1[v]); else C[r * N + c1] = acc1[v]; }
  }
}

// ---------------------------------------------------------------------------
// RoPE in place on (BT, H*RD) buffer. Pair (d, d+16) per head, angle = t*inv[d].
// ---------------------------------------------------------------------------
__global__ void rope_kernel(float* __restrict__ buf) {
  int i = blockIdx.x * blockDim.x + threadIdx.x;   // BT * H * 16
  if (i >= BT_ * H_ * 16) return;
  int d = i & 15;
  int h = (i >> 4) & (H_ - 1);
  int row = i >> 7;
  int t = row % T_;
  float inv = __expf(-(float)d * (logf(10000.f) / 16.f));  // 10000^(-d/16)
  float ang = (float)t * inv;
  float c = __cosf(ang), s = __sinf(ang);
  float* p = buf + (size_t)row * (H_ * RD_) + h * RD_;
  float x1 = p[d], x2 = p[d + 16];
  p[d]      = x1 * c - x2 * s;
  p[d + 16] = x2 * c + x1 * s;
}

// ---------------------------------------------------------------------------
// Causal attention. q = [qc(48) | qr(32)], k = [kv(48) | kr(32)], v = kv(48).
// Grid (B*H, T/64); 8 waves/block, one query row per wave per iteration.
// ---------------------------------------------------------------------------
__global__ void attn_kernel(const float* __restrict__ qc, const float* __restrict__ qr,
                            const float* __restrict__ kvb, const float* __restrict__ krb,
                            float* __restrict__ y) {
  const float scale = 0.14433756729740643f;  // 1/sqrt(48)
  int b = blockIdx.x / H_, h = blockIdx.x % H_;
  int w = threadIdx.x >> 5, lane = threadIdx.x & 31;
  __shared__ float qsh[8][80];
  __shared__ float sc[8][512];

  for (int it = 0; it < 8; ++it) {
    int tq = blockIdx.y * 64 + it * 8 + w;
    size_t row = (size_t)b * T_ + tq;
    for (int d = lane; d < 80; d += 32)
      qsh[w][d] = (d < 48) ? qc[row * (H_ * HS_) + h * HS_ + d]
                           : qr[row * (H_ * RD_) + h * RD_ + (d - 48)];
    __syncthreads();

    float mx = -1e30f;
    for (int j = lane; j <= tq; j += 32) {
      const float* kc = kvb + ((size_t)b * T_ + j) * (H_ * HS_) + h * HS_;
      const float* kr = krb + ((size_t)b * T_ + j) * (H_ * RD_) + h * RD_;
      float s = 0.f;
      #pragma unroll 8
      for (int d = 0; d < 48; ++d) s += qsh[w][d] * kc[d];
      #pragma unroll 8
      for (int d = 0; d < 32; ++d) s += qsh[w][48 + d] * kr[d];
      s *= scale;
      sc[w][j] = s;
      mx = fmaxf(mx, s);
    }
    for (int o = 16; o > 0; o >>= 1) mx = fmaxf(mx, __shfl_xor(mx, o, 32));

    float se = 0.f;
    for (int j = lane; j <= tq; j += 32) {
      float p = __expf(sc[w][j] - mx);
      sc[w][j] = p;
      se += p;
    }
    for (int o = 16; o > 0; o >>= 1) se += __shfl_xor(se, o, 32);
    __syncthreads();

    float inv = 1.f / se;
    float a0 = 0.f, a1 = 0.f;
    for (int j = 0; j <= tq; ++j) {
      float p = sc[w][j];
      const float* v = kvb + ((size_t)b * T_ + j) * (H_ * HS_) + h * HS_;
      a0 += p * v[lane];
      if (lane < 16) a1 += p * v[32 + lane];
    }
    y[row * (H_ * HS_) + h * HS_ + lane] = a0 * inv;
    if (lane < 16) y[row * (H_ * HS_) + h * HS_ + 32 + lane] = a1 * inv;
  }
}

// ---------------------------------------------------------------------------
// g = silu(a) * b, bf16 out (feeds W3 GEMM)
// ---------------------------------------------------------------------------
__global__ void silu_mul_kernel(const float* __restrict__ a, const float* __restrict__ b,
                                bf16* __restrict__ g, long n) {
  long i = (long)blockIdx.x * blockDim.x + threadIdx.x;
  if (i >= n) return;
  float av = a[i];
  g[i] = (bf16)((av / (1.f + __expf(-av))) * b[i]);
}

__global__ void zero_loss_kernel(float* loss) { loss[0] = 0.f; }

// ---------------------------------------------------------------------------
// Cross-entropy: per-row log-softmax over V, atomicAdd of -logp[target]/BT
// ---------------------------------------------------------------------------
__global__ void loss_kernel(const float* __restrict__ logits, const int* __restrict__ targets,
                            float* __restrict__ loss) {
  int row = blockIdx.x, tid = threadIdx.x;
  const float* lr = logits + (size_t)row * V_;
  __shared__ float red[256];
  float mx = -1e30f;
  for (int j = tid; j < V_; j += 256) mx = fmaxf(mx, lr[j]);
  red[tid] = mx; __syncthreads();
  for (int o = 128; o > 0; o >>= 1) { if (tid < o) red[tid] = fmaxf(red[tid], red[tid + o]); __syncthreads(); }
  mx = red[0]; __syncthreads();
  float s = 0.f;
  for (int j = tid; j < V_; j += 256) s += __expf(lr[j] - mx);
  red[tid] = s; __syncthreads();
  for (int o = 128; o > 0; o >>= 1) { if (tid < o) red[tid] += red[tid + o]; __syncthreads(); }
  if (tid == 0) {
    float lp = lr[targets[row]] - mx - logf(red[0]);
    atomicAdd(loss, -lp / (float)BT_);
  }
}

// ---------------------------------------------------------------------------
// host orchestration
// ---------------------------------------------------------------------------
extern "C" void kernel_launch(void* const* d_in, const int* in_sizes, int n_in,
                              void* d_out, int out_size, void* d_ws, size_t ws_size,
                              hipStream_t stream) {
  (void)in_sizes; (void)n_in; (void)out_size; (void)ws_size;

  const int*   idx      = (const int*)d_in[0];
  const int*   targets  = (const int*)d_in[1];
  const float* tok_emb  = (const float*)d_in[2];
  const float* pos_emb  = (const float*)d_in[3];
  const float* kv_down  = (const float*)d_in[4];
  const float* kv_up    = (const float*)d_in[5];
  const float* k_rope_w = (const float*)d_in[6];
  const float* q_down   = (const float*)d_in[7];
  const float* q_up     = (const float*)d_in[8];
  const float* q_rope_w = (const float*)d_in[9];
  const float* out_proj = (const float*)d_in[10];
  const float* w1       = (const float*)d_in[11];
  const float* w2       = (const float*)d_in[12];
  const float* w3       = (const float*)d_in[13];
  const float* ln1_g    = (const float*)d_in[14];
  const float* ln1_b    = (const float*)d_in[15];
  const float* ln2_g    = (const float*)d_in[16];
  const float* ln2_b    = (const float*)d_in[17];
  const float* lnf_g    = (const float*)d_in[18];
  const float* lnf_b    = (const float*)d_in[19];

  float* logits = (float*)d_out;
  float* loss   = logits + (size_t)BT_ * V_;

  // ---- carve workspace ----
  char* ws = (char*)d_ws;
  size_t off = 0;
  auto carve = [&](size_t bytes) -> void* {
    void* p = (void*)(ws + off);
    off += (bytes + 255) & ~(size_t)255;
    return p;
  };
  // weights, bf16, all stored TRANSPOSED as [N][K] per layer (tok_emb already N x K)
  bf16* tokemb16 = (bf16*)carve((size_t)V_ * E_ * 2);
  bf16* kvd16    = (bf16*)carve((size_t)L_ * E_ * KVR_ * 2);
  bf16* kvu16    = (bf16*)carve((size_t)L_ * KVR_ * (H_*HS_) * 2);
  bf16* krw16    = (bf16*)carve((size_t)L_ * KVR_ * (H_*RD_) * 2);
  bf16* qd16     = (bf16*)carve((size_t)L_ * E_ * QR_ * 2);
  bf16* qu16     = (bf16*)carve((size_t)L_ * QR_ * (H_*HS_) * 2);
  bf16* qrw16    = (bf16*)carve((size_t)L_ * QR_ * (H_*RD_) * 2);
  bf16* op16     = (bf16*)carve((size_t)L_ * (H_*HS_) * E_ * 2);
  bf16* w1_16    = (bf16*)carve((size_t)L_ * E_ * HID_ * 2);
  bf16* w2_16    = (bf16*)carve((size_t)L_ * E_ * HID_ * 2);
  bf16* w3_16    = (bf16*)carve((size_t)L_ * HID_ * E_ * 2);

  float* x     = (float*)carve((size_t)BT_ * E_ * 4);
  bf16*  h16   = (bf16*) carve((size_t)BT_ * E_ * 2);
  float* lkv   = (float*)carve((size_t)BT_ * KVR_ * 4);
  bf16*  lkv16 = (bf16*) carve((size_t)BT_ * KVR_ * 2);
  float* lq    = (float*)carve((size_t)BT_ * QR_ * 4);
  bf16*  lq16  = (bf16*) carve((size_t)BT_ * QR_ * 2);
  float* kvb   = (float*)carve((size_t)BT_ * (H_*HS_) * 4);   // kc and v
  float* krb   = (float*)carve((size_t)BT_ * (H_*RD_) * 4);
  float* qcb   = (float*)carve((size_t)BT_ * (H_*HS_) * 4);
  float* qrb   = (float*)carve((size_t)BT_ * (H_*RD_) * 4);
  float* yb    = (float*)carve((size_t)BT_ * (H_*HS_) * 4);
  bf16*  y16   = (bf16*) carve((size_t)BT_ * (H_*HS_) * 2);
  float* mlpa  = (float*)carve((size_t)BT_ * HID_ * 4);
  float* mlpb  = (float*)carve((size_t)BT_ * HID_ * 4);
  bf16*  g16   = (bf16*) carve((size_t)BT_ * HID_ * 2);

  auto conv = [&](const float* a, bf16* o, long n) {
    f32_to_bf16_kernel<<<dim3((unsigned)((n + 255) / 256)), 256, 0, stream>>>(a, o, n);
  };
  // convert+transpose one [L][K][N] weight -> [L][N][K] bf16
  auto convT = [&](const float* a, bf16* o, int K, int N) {
    transpose_f32_to_bf16_kernel<<<dim3(N / 32, K / 32, L_), 256, 0, stream>>>(a, o, K, N);
  };
  auto gemm = [&](const bf16* A, const bf16* Bt, float* C, int M, int N, int K, bool acc) {
    dim3 g((N + 63) / 64, (M + 63) / 64);
    if (acc) gemm_wmma<true ><<<g, 256, 0, stream>>>(A, Bt, C, M, N, K);
    else     gemm_wmma<false><<<g, 256, 0, stream>>>(A, Bt, C, M, N, K);
  };

  // ---- weights -> bf16, transposed (every call; deterministic) ----
  conv (tok_emb,  tokemb16, (long)V_ * E_);          // already [V][E] = [N][K]
  convT(kv_down,  kvd16,  E_,   KVR_);
  convT(kv_up,    kvu16,  KVR_, H_*HS_);
  convT(k_rope_w, krw16,  KVR_, H_*RD_);
  convT(q_down,   qd16,   E_,   QR_);
  convT(q_up,     qu16,   QR_,  H_*HS_);
  convT(q_rope_w, qrw16,  QR_,  H_*RD_);
  convT(out_proj, op16,   H_*HS_, E_);
  convT(w1,       w1_16,  E_,   HID_);
  convT(w2,       w2_16,  E_,   HID_);
  convT(w3,       w3_16,  HID_, E_);

  // ---- embedding ----
  embed_kernel<<<dim3((BT_ * E_ + 255) / 256), 256, 0, stream>>>(idx, tok_emb, pos_emb, x);

  // ---- transformer layers ----
  for (int l = 0; l < L_; ++l) {
    ln_kernel<<<BT_, 256, 0, stream>>>(x, ln1_g + (size_t)l * E_, ln1_b + (size_t)l * E_, h16);

    gemm(h16, kvd16 + (size_t)l * E_ * KVR_, lkv, BT_, KVR_, E_, false);
    conv(lkv, lkv16, (long)BT_ * KVR_);
    gemm(lkv16, kvu16 + (size_t)l * KVR_ * (H_*HS_), kvb, BT_, H_*HS_, KVR_, false);
    gemm(lkv16, krw16 + (size_t)l * KVR_ * (H_*RD_), krb, BT_, H_*RD_, KVR_, false);

    gemm(h16, qd16 + (size_t)l * E_ * QR_, lq, BT_, QR_, E_, false);
    conv(lq, lq16, (long)BT_ * QR_);
    gemm(lq16, qu16 + (size_t)l * QR_ * (H_*HS_), qcb, BT_, H_*HS_, QR_, false);
    gemm(lq16, qrw16 + (size_t)l * QR_ * (H_*RD_), qrb, BT_, H_*RD_, QR_, false);

    rope_kernel<<<dim3((BT_ * H_ * 16 + 255) / 256), 256, 0, stream>>>(krb);
    rope_kernel<<<dim3((BT_ * H_ * 16 + 255) / 256), 256, 0, stream>>>(qrb);

    attn_kernel<<<dim3(B_ * H_, T_ / 64), 256, 0, stream>>>(qcb, qrb, kvb, krb, yb);

    conv(yb, y16, (long)BT_ * (H_*HS_));
    gemm(y16, op16 + (size_t)l * (H_*HS_) * E_, x, BT_, E_, H_*HS_, true);   // x += y @ out_proj

    ln_kernel<<<BT_, 256, 0, stream>>>(x, ln2_g + (size_t)l * E_, ln2_b + (size_t)l * E_, h16);
    gemm(h16, w1_16 + (size_t)l * E_ * HID_, mlpa, BT_, HID_, E_, false);
    gemm(h16, w2_16 + (size_t)l * E_ * HID_, mlpb, BT_, HID_, E_, false);
    silu_mul_kernel<<<dim3((BT_ * HID_ + 255) / 256), 256, 0, stream>>>(mlpa, mlpb, g16, (long)BT_ * HID_);
    gemm(g16, w3_16 + (size_t)l * HID_ * E_, x, BT_, E_, HID_, true);        // x += mlp
  }

  // ---- final LN + tied logits GEMM (B = tok_emb, already [N][K]) + loss ----
  ln_kernel<<<BT_, 256, 0, stream>>>(x, lnf_g, lnf_b, h16);
  {
    dim3 g((V_ + 63) / 64, (BT_ + 63) / 64);
    gemm_wmma<false><<<g, 256, 0, stream>>>(h16, tokemb16, logits, BT_, V_, E_);
  }
  zero_loss_kernel<<<1, 1, 0, stream>>>(loss);
  loss_kernel<<<BT_, 256, 0, stream>>>(logits, targets, loss);
}
